// SMCSampler1_8229157339136
// MI455X (gfx1250) — compile-verified, matrix operand order checked by reading, not run on hardware
//
#include <hip/hip_runtime.h>
#include <hip/hip_bf16.h>
#include <math.h>

// MI455X / gfx1250 (wave32). SMC sampler with MALA propagation.
// Design: latency/VALU-bound workload -> register-resident particle state,
// hash-based stateless RNG (graph-replay deterministic), WMMA used for the
// genuinely-matrix part (X @ [mu_{t-1}, mu_t] dot products in the reweight).

typedef __attribute__((ext_vector_type(16))) _Float16 v16h;
typedef __attribute__((ext_vector_type(8)))  float    v8f;

#define WGSIZE 256

// ---------------- stateless RNG ----------------
__device__ __forceinline__ unsigned hash_u32(unsigned x) {
  x ^= x >> 16; x *= 0x7feb352du;
  x ^= x >> 15; x *= 0x846ca68bu;
  x ^= x >> 16; return x;
}
__device__ __forceinline__ float u01(unsigned h) {
  return (float)(h >> 8) * (1.0f / 16777216.0f);   // [0,1)
}

// ---------------- fused 50-step MALA chain ----------------
// For isotropic Gaussian target: y=x-mu, a=1-tau/sig^2,
// y' = a*y + sqrt(2 tau)*xi,
// acc_lp = (|y|^2-|y'|^2)/(2 sig^2) - |y - a*y'|^2/(4 tau) + |xi|^2/2.
__global__ void mala_kernel(const float* __restrict__ xin, float* __restrict__ xout,
                            const float* __restrict__ mus, const float* __restrict__ sigs,
                            int stage, int n, unsigned salt, int K) {
  int p = blockIdx.x * blockDim.x + threadIdx.x;
  if (p >= n) return;
  const float* mu = mus + stage * 8;
  float sig = sigs[stage];
  float inv_s2 = 1.0f / (sig * sig);

  const float4* src = (const float4*)(xin + (size_t)p * 8);
  __builtin_prefetch(src, 0, 1);                 // global_prefetch_b8
  float4 lo = src[0], hi = src[1];
  float y[8] = { lo.x - mu[0], lo.y - mu[1], lo.z - mu[2], lo.w - mu[3],
                 hi.x - mu[4], hi.y - mu[5], hi.z - mu[6], hi.w - mu[7] };
  float ssY = 0.0f;
#pragma unroll
  for (int d = 0; d < 8; ++d) ssY = fmaf(y[d], y[d], ssY);

  for (int k = 1; k <= K; ++k) {
    float tau = 1.0f / (float)k;
    float a   = 1.0f - tau * inv_s2;
    float sn  = sqrtf(2.0f * tau);
    unsigned base = hash_u32((unsigned)p * 0x9E3779B9u ^ (unsigned)k * 0x85EBCA6Bu ^ salt);

    float xi[8];
#pragma unroll
    for (int j = 0; j < 4; ++j) {                // Box-Muller pairs
      float u1 = fmaxf(u01(hash_u32(base + (unsigned)(2*j)   * 0x632BE5ABu)), 1e-7f);
      float u2 =       u01(hash_u32(base + (unsigned)(2*j+1) * 0x632BE5ABu));
      float r  = sqrtf(-2.0f * __logf(u1));
      float sv, cv; __sincosf(6.28318530718f * u2, &sv, &cv);
      xi[2*j]   = r * cv;
      xi[2*j+1] = r * sv;
    }
    float uac = u01(hash_u32(base + 8u * 0x632BE5ABu));

    float yn[8]; float ssYn = 0.0f, ssD = 0.0f, ssXi = 0.0f;
#pragma unroll
    for (int d = 0; d < 8; ++d) {
      float v  = fmaf(a, y[d], sn * xi[d]);
      yn[d] = v;
      ssYn = fmaf(v, v, ssYn);
      float df = fmaf(-a, v, y[d]);
      ssD  = fmaf(df, df, ssD);
      ssXi = fmaf(xi[d], xi[d], ssXi);
    }
    float acc = (ssY - ssYn) * (0.5f * inv_s2) - ssD * (0.25f / tau) + 0.5f * ssXi;
    bool ok = uac < __expf(acc);
#pragma unroll
    for (int d = 0; d < 8; ++d) y[d] = ok ? yn[d] : y[d];
    ssY = ok ? ssYn : ssY;
  }

  float4 o0 = { y[0]+mu[0], y[1]+mu[1], y[2]+mu[2], y[3]+mu[3] };
  float4 o1 = { y[4]+mu[4], y[5]+mu[5], y[6]+mu[6], y[7]+mu[7] };
  float4* dst = (float4*)(xout + (size_t)p * 8);
  dst[0] = o0; dst[1] = o1;
}

// ---------------- reductions / scans for multinomial resampling ----------------
__global__ void rmax_part(const float* __restrict__ v, float* __restrict__ part, int n) {
  __shared__ float sm[WGSIZE];
  int i = blockIdx.x * blockDim.x + threadIdx.x;
  sm[threadIdx.x] = (i < n) ? v[i] : -3.4e38f;
  __syncthreads();
  for (int off = WGSIZE/2; off > 0; off >>= 1) {
    if (threadIdx.x < off) sm[threadIdx.x] = fmaxf(sm[threadIdx.x], sm[threadIdx.x + off]);
    __syncthreads();
  }
  if (threadIdx.x == 0) part[blockIdx.x] = sm[0];
}

__global__ void rmax_final(const float* __restrict__ part, float* __restrict__ rstat, int nb) {
  __shared__ float sm[WGSIZE];
  sm[threadIdx.x] = (threadIdx.x < nb) ? part[threadIdx.x] : -3.4e38f;
  __syncthreads();
  for (int off = WGSIZE/2; off > 0; off >>= 1) {
    if (threadIdx.x < off) sm[threadIdx.x] = fmaxf(sm[threadIdx.x], sm[threadIdx.x + off]);
    __syncthreads();
  }
  if (threadIdx.x == 0) rstat[0] = sm[0];
}

// Per-block inclusive scan of weights (mode 0: raw w0; mode 1: exp(lw - max)).
__global__ void scan_block(const float* __restrict__ lw, const float* __restrict__ w0,
                           const float* __restrict__ rstat, int mode,
                           float* __restrict__ cdf, float* __restrict__ bsum, int n) {
  __shared__ float sm[WGSIZE];
  int i = blockIdx.x * WGSIZE + threadIdx.x;
  float w = 0.0f;
  if (i < n) w = mode ? __expf(lw[i] - rstat[0]) : w0[i];
  sm[threadIdx.x] = w;
  __syncthreads();
  for (int off = 1; off < WGSIZE; off <<= 1) {       // Hillis-Steele inclusive
    float v = (threadIdx.x >= off) ? sm[threadIdx.x - off] : 0.0f;
    __syncthreads();
    sm[threadIdx.x] += v;
    __syncthreads();
  }
  if (i < n) cdf[i] = sm[threadIdx.x];
  if (threadIdx.x == WGSIZE - 1) bsum[blockIdx.x] = sm[WGSIZE - 1];
}

__global__ void scan_bsums(const float* __restrict__ bsum, float* __restrict__ bpre, int nb) {
  __shared__ float sm[WGSIZE];
  sm[threadIdx.x] = (threadIdx.x < nb) ? bsum[threadIdx.x] : 0.0f;
  __syncthreads();
  for (int off = 1; off < WGSIZE; off <<= 1) {
    float v = (threadIdx.x >= off) ? sm[threadIdx.x - off] : 0.0f;
    __syncthreads();
    sm[threadIdx.x] += v;
    __syncthreads();
  }
  if (threadIdx.x < nb) bpre[threadIdx.x] = sm[threadIdx.x];
}

// Inverse-CDF categorical sample + 8-float gather.
__global__ void gather_kernel(const float* __restrict__ prop, float* __restrict__ dst,
                              const float* __restrict__ cdf, const float* __restrict__ bpre,
                              int n, int nb, unsigned salt) {
  int j = blockIdx.x * blockDim.x + threadIdx.x;
  if (j >= n) return;
  float total = bpre[nb - 1];
  float u = u01(hash_u32((unsigned)j * 0xB5297A4Du ^ salt)) * total;
  int lo = 0, hi = nb - 1;                            // upper_bound over block prefixes
  while (lo < hi) { int mid = (lo + hi) >> 1; if (u < bpre[mid]) hi = mid; else lo = mid + 1; }
  int b = lo;
  float uin = u - ((b > 0) ? bpre[b - 1] : 0.0f);
  const float* c = cdf + b * WGSIZE;
  int l2 = 0, h2 = WGSIZE - 1;                        // upper_bound inside block
  while (l2 < h2) { int mid = (l2 + h2) >> 1; if (uin < c[mid]) h2 = mid; else l2 = mid + 1; }
  int idx = b * WGSIZE + l2;
  const float4* s = (const float4*)(prop + (size_t)idx * 8);
  float4* d = (float4*)(dst + (size_t)j * 8);
  d[0] = s[0]; d[1] = s[1];
}

// ---------------- incremental weights via WMMA ----------------
// ulw = ld_t(x) - ln N - ld_{t-1}(x),  |x-mu|^2 = |x|^2 - 2<x,mu> + |mu|^2.
// The <x,mu> cross terms for a 16-particle tile are a 16x8 * 8x2 GEMM:
// one v_wmma_f32_16x16x32_f16 with K zero-padded (cols 0/1 = mu_{t-1}, mu_t).
// A layout: lanes 0..15 carry K=0..7 of row m in f16 elements 0..7 (rest 0).
// Dot results land in D columns 0/1 -> lanes {0,1,16,17}; routed via LDS.
__global__ void ulw_wmma_kernel(const float* __restrict__ x, float* __restrict__ lw,
                                const float* __restrict__ mus, const float* __restrict__ sigs,
                                int stage, int n) {
  __shared__ float dots[8][2][16];   // [wave][col][row]
  int lane = threadIdx.x & 31;
  int wave = threadIdx.x >> 5;
  int p = (blockIdx.x * 8 + wave) * 16 + (lane & 15);
  const float* mu0 = mus + stage * 8;
  const float* mu1 = mus + (stage + 1) * 8;

  v16h a, b;
#pragma unroll
  for (int k = 0; k < 16; ++k) { a[k] = (_Float16)0.0f; b[k] = (_Float16)0.0f; }

  float q = 0.0f;
  if (lane < 16 && p < n) {
    const float4* s = (const float4*)(x + (size_t)p * 8);
    float4 lo = s[0], hi = s[1];
    float xv[8] = { lo.x, lo.y, lo.z, lo.w, hi.x, hi.y, hi.z, hi.w };
#pragma unroll
    for (int k = 0; k < 8; ++k) { q = fmaf(xv[k], xv[k], q); a[k] = (_Float16)xv[k]; }
  }
  if (lane == 0) {
#pragma unroll
    for (int k = 0; k < 8; ++k) b[k] = (_Float16)mu0[k];
  } else if (lane == 1) {
#pragma unroll
    for (int k = 0; k < 8; ++k) b[k] = (_Float16)mu1[k];
  }

  v8f c = {};
  // full-EXEC point (divergent ifs above have reconverged)
  v8f d = __builtin_amdgcn_wmma_f32_16x16x32_f16(false, a, false, b,
                                                 (short)0, c, false, false);

  if (lane == 0 || lane == 1 || lane == 16 || lane == 17) {
    int col = lane & 1;
    int rb  = (lane >= 16) ? 8 : 0;
#pragma unroll
    for (int r = 0; r < 8; ++r) dots[wave][col][rb + r] = d[r];
  }
  __syncthreads();

  if (lane < 16 && p < n) {
    float dot0 = dots[wave][0][lane & 15];
    float dot1 = dots[wave][1][lane & 15];
    float nm0 = 0.0f, nm1 = 0.0f;
#pragma unroll
    for (int k = 0; k < 8; ++k) {
      nm0 = fmaf(mu0[k], mu0[k], nm0);
      nm1 = fmaf(mu1[k], mu1[k], nm1);
    }
    float s0 = sigs[stage], s1 = sigs[stage + 1];
    float i0 = 1.0f / (s0 * s0), i1 = 1.0f / (s1 * s1);
    const float L2PI = 1.8378770664093453f;   // ln(2*pi)
    float ld1 = -0.5f * (q - 2.0f * dot1 + nm1) * i1 - 4.0f * (L2PI + 2.0f * __logf(s1));
    float ld0 = -0.5f * (q - 2.0f * dot0 + nm0) * i0 - 4.0f * (L2PI + 2.0f * __logf(s0));
    lw[p] = ld1 - __logf((float)n) - ld0;
  }
}

// ---------------- host orchestration ----------------
extern "C" void kernel_launch(void* const* d_in, const int* in_sizes, int n_in,
                              void* d_out, int out_size, void* d_ws, size_t ws_size,
                              hipStream_t stream) {
  (void)n_in; (void)out_size; (void)ws_size;
  const float* x0   = (const float*)d_in[0];   // [N,8]
  const float* w0   = (const float*)d_in[1];   // [N] normalized
  const float* mus  = (const float*)d_in[2];   // [T,8]
  const float* sigs = (const float*)d_in[3];   // [T]
  int N  = in_sizes[1];
  int T  = in_sizes[3];
  int NB = N / WGSIZE;

  char* ws = (char*)d_ws;
  size_t off = 0;
  float* bufA  = (float*)(ws + off); off += (size_t)N * 8 * sizeof(float);
  float* bufB  = (float*)(ws + off); off += (size_t)N * 8 * sizeof(float);
  float* lw    = (float*)(ws + off); off += (size_t)N * sizeof(float);
  float* cdf   = (float*)(ws + off); off += (size_t)N * sizeof(float);
  float* bsum  = (float*)(ws + off); off += 1024;
  float* bpre  = (float*)(ws + off); off += 1024;
  float* part  = (float*)(ws + off); off += 1024;
  float* rstat = (float*)(ws + off);

  dim3 blk(WGSIZE);
  for (int s = 0; s <= T - 2; ++s) {
    const float* xc = (s == 0) ? x0 : bufB;
    float* dst = (s == T - 2) ? (float*)d_out : bufB;

    // 1) propagate: 50-step MALA chain targeting density s
    mala_kernel<<<dim3(N / WGSIZE), blk, 0, stream>>>(
        xc, bufA, mus, sigs, s, N, 0xA3417C2Du * (unsigned)(s + 1), 50);

    // 2) resample with previous-stage weights (unnormalized lw suffices)
    if (s > 0) {
      rmax_part<<<dim3(NB), blk, 0, stream>>>(lw, part, N);
      rmax_final<<<dim3(1), blk, 0, stream>>>(part, rstat, NB);
    }
    scan_block<<<dim3(NB), blk, 0, stream>>>(lw, w0, rstat, (s > 0) ? 1 : 0, cdf, bsum, N);
    scan_bsums<<<dim3(1), blk, 0, stream>>>(bsum, bpre, NB);
    gather_kernel<<<dim3(N / WGSIZE), blk, 0, stream>>>(
        bufA, dst, cdf, bpre, N, NB, 0x5F1E9C3Bu * (unsigned)(s + 7));

    // 3) reweight (skipped on final stage: logw unused by the reference output)
    if (s < T - 2)
      ulw_wmma_kernel<<<dim3(N / 128), blk, 0, stream>>>(dst, lw, mus, sigs, s, N);
  }
}
